// CorefScore_910533066866
// MI455X (gfx1250) — matrix operand order repaired and checked by reading, not compiled, not run on hardware
//
#include <hip/hip_runtime.h>
#include <hip/hip_bf16.h>
#include <math.h>

typedef __attribute__((ext_vector_type(16))) _Float16 v16h;
typedef __attribute__((ext_vector_type(8)))  _Float16 v8h;
typedef __attribute__((ext_vector_type(8)))  float    v8f;

#define TT     1024
#define EE     400
#define HH     200
#define G4H    800
#define SS     8192
#define WMAX   8
#define MM     300
#define KK     100
#define HS     150
#define DD     1220
#define PP     3720
#define MK     (MM*KK)      // 30000
#define MKP    30016        // MK padded to 32
#define KIN    416          // 400 padded to 32
#define KD     1248         // 1220 padded to 32
#define KH     160          // 150 padded to 32
#define MM2    320          // 300 padded to 32

__device__ __forceinline__ float sigm(float x) { return 1.0f / (1.0f + expf(-x)); }

__device__ __forceinline__ int binize(int n) {
    int r = 0;
    r += (n >= 1);  r += (n >= 2);  r += (n >= 3);  r += (n >= 4);
    r += (n >= 8);  r += (n >= 16); r += (n >= 32); r += (n >= 64);
    return r;
}

// A fragment (16x32, lane = row, halves split K 0-7/16-23 vs 8-15/24-31)
__device__ __forceinline__ v16h load_fragA(const _Float16* __restrict__ row,
                                           int k0, int ka) {
    v8h x0 = *(const v8h*)(row + k0 + ka);
    v8h x1 = *(const v8h*)(row + k0 + ka + 16);
    return __builtin_shufflevector(x0, x1, 0,1,2,3,4,5,6,7,8,9,10,11,12,13,14,15);
}
// B fragment (32x16, lane = col, halves split K 0-7/16-23 vs 8-15/24-31)
__device__ __forceinline__ v16h load_fragB(const _Float16* __restrict__ row,
                                           int k0, int kb) {
    v8h x0 = *(const v8h*)(row + k0 + kb);
    v8h x1 = *(const v8h*)(row + k0 + kb + 16);
    return __builtin_shufflevector(x0, x1, 0,1,2,3,4,5,6,7,8,9,10,11,12,13,14,15);
}

// fp32 -> padded f16 pack (rows -> Rp, cols -> Cp, zero fill)
__global__ void pack_f16_kernel(const float* __restrict__ src, int rows, int cols,
                                int ld, _Float16* __restrict__ dst, int Rp, int Cp)
{
    int idx = blockIdx.x * blockDim.x + threadIdx.x;
    if (idx >= Rp * Cp) return;
    int r = idx / Cp, c = idx - r * Cp;
    float v = (r < rows && c < cols) ? src[(size_t)r * ld + c] : 0.0f;
    dst[idx] = (_Float16)v;
}

// WMMA GEMM, 2x2 register-blocked: each wave computes a 32x32 tile with 4
// accumulators; 8 b128 loads feed 4 v_wmma per 32-K step (2 loads/WMMA).
__global__ __launch_bounds__(32) void gemm_pk_kernel(
    const _Float16* __restrict__ A, int ldaP,
    const _Float16* __restrict__ W, int ldwP,
    const float* __restrict__ bias,
    int Kp, int Mr, int Nc,
    float* __restrict__ Cf, int ldc,
    _Float16* __restrict__ Ch, int ldcP,
    int relu)
{
    const int lane = threadIdx.x & 31;
    const int m0 = blockIdx.x * 32;
    const int n0 = blockIdx.y * 32;
    const int ka = (lane >> 4) * 8;
    const int kb = (lane >> 4) * 8;

    const _Float16* Ar0 = A + (size_t)(m0 + (lane & 15)) * ldaP;
    const _Float16* Ar1 = Ar0 + (size_t)16 * ldaP;
    const _Float16* Wr0 = W + (size_t)(n0 + (lane & 15)) * ldwP;
    const _Float16* Wr1 = Wr0 + (size_t)16 * ldwP;

    v8f acc[2][2] = {};
    for (int k0 = 0; k0 < Kp; k0 += 32) {
        __builtin_prefetch(Ar0 + k0 + 512, 0, 1);
        __builtin_prefetch(Ar1 + k0 + 512, 0, 1);
        __builtin_prefetch(Wr0 + k0 + 512, 0, 1);
        __builtin_prefetch(Wr1 + k0 + 512, 0, 1);
        v16h a0 = load_fragA(Ar0, k0, ka);
        v16h a1 = load_fragA(Ar1, k0, ka);
        v16h b0 = load_fragB(Wr0, k0, kb);
        v16h b1 = load_fragB(Wr1, k0, kb);
        acc[0][0] = __builtin_amdgcn_wmma_f32_16x16x32_f16(false, a0, false, b0, (short)0, acc[0][0], false, false);
        acc[0][1] = __builtin_amdgcn_wmma_f32_16x16x32_f16(false, a0, false, b1, (short)0, acc[0][1], false, false);
        acc[1][0] = __builtin_amdgcn_wmma_f32_16x16x32_f16(false, a1, false, b0, (short)0, acc[1][0], false, false);
        acc[1][1] = __builtin_amdgcn_wmma_f32_16x16x32_f16(false, a1, false, b1, (short)0, acc[1][1], false, false);
    }

#pragma unroll
    for (int rf = 0; rf < 2; ++rf) {
        const int mbase = m0 + rf * 16 + ((lane >> 4) << 3);
#pragma unroll
        for (int cf = 0; cf < 2; ++cf) {
            const int col = n0 + cf * 16 + (lane & 15);
            const float bv = (bias && col < Nc) ? bias[col] : 0.0f;
#pragma unroll
            for (int v = 0; v < 8; ++v) {
                int m = mbase + v;
                float r = acc[rf][cf][v] + bv;
                if (relu) r = fmaxf(r, 0.0f);
                if (Cf) {
                    if (m < Mr && col < Nc) Cf[(size_t)m * ldc + col] = r;
                }
                if (Ch) {
                    Ch[(size_t)m * ldcP + col] = (col < Nc) ? (_Float16)r : (_Float16)0.0f;
                }
            }
        }
    }
}

// biLSTM recurrence (sequential); blockIdx.x = direction; float4 weight loads.
__global__ __launch_bounds__(G4H) void lstm_scan_kernel(
    const float* __restrict__ pre0, const float* __restrict__ pre1,
    const float* __restrict__ whh0, const float* __restrict__ whh1,
    float* __restrict__ out)
{
    const int dir = blockIdx.x;
    const float* pre = dir ? pre1 : pre0;
    const float* whh = dir ? whh1 : whh0;

    __shared__ float h[HH], c[HH], gsh[G4H];
    const int j = threadIdx.x;
    if (j < HH) { h[j] = 0.0f; c[j] = 0.0f; }
    __syncthreads();

    const float4* w4 = (const float4*)(whh + (size_t)j * HH);
    for (int t = 0; t < TT; ++t) {
        const int r = dir ? (TT - 1 - t) : t;
        {
            float s = pre[(size_t)r * G4H + j];
#pragma unroll 10
            for (int q = 0; q < HH / 4; ++q) {
                float4 w = w4[q];
                int k4 = q * 4;
                s += h[k4] * w.x + h[k4 + 1] * w.y + h[k4 + 2] * w.z + h[k4 + 3] * w.w;
            }
            gsh[j] = s;
        }
        __syncthreads();
        if (j < HH) {
            float ig = gsh[j], fg = gsh[HH + j], gg = gsh[2 * HH + j], og = gsh[3 * HH + j];
            float cn = sigm(fg) * c[j] + sigm(ig) * tanhf(gg);
            float hn = sigm(og) * tanhf(cn);
            c[j] = cn; h[j] = hn;
            out[(size_t)r * (2 * HH) + dir * HH + j] = hn;
        }
        __syncthreads();
    }
}

// Span features: g[s] = [states[st], states[en], softmax-pooled x, width emb]
__global__ __launch_bounds__(128) void span_kernel(
    const float* __restrict__ x, const float* __restrict__ states,
    const float* __restrict__ attn,
    const int* __restrict__ starts, const int* __restrict__ ends,
    const float* __restrict__ width_tab, float* __restrict__ g)
{
    const int s = blockIdx.x;
    const int st = starts[s], en = ends[s];
    __shared__ float aw[WMAX];

    if (threadIdx.x == 0) {
        float av[WMAX]; int vmask[WMAX];
        float mx = -3.0e38f;
        for (int w = 0; w < WMAX; ++w) {
            int t = st + w;
            vmask[w] = (t <= en);
            int tc = t < (TT - 1) ? t : (TT - 1);
            av[w] = attn[tc];
            if (vmask[w] && av[w] > mx) mx = av[w];
        }
        float sum = 0.0f;
        for (int w = 0; w < WMAX; ++w) {
            float e = vmask[w] ? expf(av[w] - mx) : 0.0f;
            aw[w] = e; sum += e;
        }
        float inv = 1.0f / sum;
        for (int w = 0; w < WMAX; ++w) aw[w] *= inv;
    }
    __syncthreads();

    float* grow = g + (size_t)s * DD;
    for (int d = threadIdx.x; d < EE; d += 128) {
        grow[d]      = states[(size_t)st * EE + d];
        grow[EE + d] = states[(size_t)en * EE + d];
        float p = 0.0f;
#pragma unroll
        for (int w = 0; w < WMAX; ++w) {
            int tc = st + w; tc = tc < (TT - 1) ? tc : (TT - 1);
            p += aw[w] * x[(size_t)tc * EE + d];
        }
        grow[2 * EE + d] = p;
    }
    if (threadIdx.x < 20) {
        int b = binize(en - st + 1);
        grow[3 * EE + threadIdx.x] = width_tab[b * 20 + threadIdx.x];
    }
}

// Deterministic top-M (lower index wins ties), compact-by-index == sorted keep.
__global__ void rank_kernel(const float* __restrict__ sm, int* __restrict__ flag)
{
    int s = blockIdx.x * blockDim.x + threadIdx.x;
    if (s >= SS) return;
    float v = sm[s];
    int r = 0;
    for (int i = 0; i < SS; ++i) {
        float u = sm[i];
        r += (u > v) || (u == v && i < s);
    }
    flag[s] = (r < MM) ? 1 : 0;
}

__global__ void compact_kernel(const int* __restrict__ flag, int* __restrict__ keep)
{
    int s = blockIdx.x * blockDim.x + threadIdx.x;
    if (s >= SS || !flag[s]) return;
    int p = 0;
    for (int i = 0; i < s; ++i) p += flag[i];
    keep[p] = s;
}

// Gather kept mentions directly into padded packed-f16 layout (MM2 x KD).
__global__ __launch_bounds__(128) void gather_kernel(
    const float* __restrict__ g, const float* __restrict__ sm,
    const int* __restrict__ starts, const int* __restrict__ ends,
    const int* __restrict__ speaker_ids, const int* __restrict__ keep,
    _Float16* __restrict__ gkP, float* __restrict__ skv,
    int* __restrict__ stk, int* __restrict__ enk, int* __restrict__ spkk)
{
    const int p = blockIdx.x;
    _Float16* dst = gkP + (size_t)p * KD;
    if (p >= MM) {
        for (int d = threadIdx.x; d < KD; d += 128) dst[d] = (_Float16)0.0f;
        if (threadIdx.x == 0) { stk[p] = 0; enk[p] = 0; spkk[p] = 0; }
        return;
    }
    const int s = keep[p];
    for (int d = threadIdx.x; d < KD; d += 128)
        dst[d] = (d < DD) ? (_Float16)g[(size_t)s * DD + d] : (_Float16)0.0f;
    if (threadIdx.x == 0) {
        skv[p] = sm[s];
        int st = starts[s];
        stk[p] = st; enk[p] = ends[s];
        int sc = st < 0 ? 0 : (st > TT - 1 ? TT - 1 : st);
        spkk[p] = speaker_ids[sc];
    }
}

// Precompute meta contributions through w1m = pair_w1[:,3660:3720].
__global__ void meta_tab_kernel(
    const float* __restrict__ pw1, const float* __restrict__ dist_tab,
    const float* __restrict__ genre_tab, const int* __restrict__ genre_p,
    const float* __restrict__ speaker_tab,
    float* __restrict__ distC, float* __restrict__ genC, float* __restrict__ spkC)
{
    int n = blockIdx.x * blockDim.x + threadIdx.x;
    if (n >= HS) return;
    const float* wm = pw1 + (size_t)n * PP + 3 * DD;
    for (int b = 0; b < 9; ++b) {
        float s = 0.0f;
        for (int e = 0; e < 20; ++e) s += dist_tab[b * 20 + e] * wm[e];
        distC[b * HS + n] = s;
    }
    {
        int gnr = *genre_p;
        float s = 0.0f;
        for (int e = 0; e < 20; ++e) s += genre_tab[gnr * 20 + e] * wm[20 + e];
        genC[n] = s;
    }
    for (int l = 0; l < 3; ++l) {
        float s = 0.0f;
        for (int e = 0; e < 20; ++e) s += speaker_tab[l * 20 + e] * wm[40 + e];
        spkC[l * HS + n] = s;
    }
}

// Fused pair layer-1 (MKP x 1248 x 150), 2x2 blocked: A fragments synthesized
// as gkP[m]*gkP[j] via packed f16 multiplies; meta folded into the epilogue.
__global__ __launch_bounds__(32) void pair_h1_kernel(
    const _Float16* __restrict__ gkP, const _Float16* __restrict__ w1pP,
    const float* __restrict__ pb1,
    const float* __restrict__ t_i, const float* __restrict__ t_j,
    const int* __restrict__ stk, const int* __restrict__ enk,
    const int* __restrict__ spkk,
    const float* __restrict__ distC, const float* __restrict__ genC,
    const float* __restrict__ spkC,
    _Float16* __restrict__ h1P)
{
    const int lane = threadIdx.x & 31;
    const int m0 = blockIdx.x * 32;
    const int n0 = blockIdx.y * 32;
    const int ka = (lane >> 4) * 8;
    const int kb = (lane >> 4) * 8;

    const _Float16* gm[2];
    const _Float16* gj[2];
#pragma unroll
    for (int rf = 0; rf < 2; ++rf) {
        int mk_row = m0 + rf * 16 + (lane & 15);
        int mmv = mk_row / KK;
        int kkv = mk_row - mmv * KK;
        int jjv = mmv - KK + kkv; if (jjv < 0) jjv = 0;
        gm[rf] = gkP + (size_t)mmv * KD;
        gj[rf] = gkP + (size_t)jjv * KD;
    }
    const _Float16* Wr0 = w1pP + (size_t)(n0 + (lane & 15)) * KD;
    const _Float16* Wr1 = Wr0 + (size_t)16 * KD;

    v8f acc[2][2] = {};
    for (int k0 = 0; k0 < KD; k0 += 32) {
        __builtin_prefetch(gm[0] + k0 + 512, 0, 1);
        __builtin_prefetch(gj[0] + k0 + 512, 0, 1);
        __builtin_prefetch(Wr0 + k0 + 512, 0, 1);
        __builtin_prefetch(Wr1 + k0 + 512, 0, 1);
        v16h a0, a1;
        {
            v8h m00 = *(const v8h*)(gm[0] + k0 + ka);
            v8h m01 = *(const v8h*)(gm[0] + k0 + ka + 16);
            v8h j00 = *(const v8h*)(gj[0] + k0 + ka);
            v8h j01 = *(const v8h*)(gj[0] + k0 + ka + 16);
            v8h p0 = m00 * j00;              // v_pk_mul_f16
            v8h p1 = m01 * j01;
            a0 = __builtin_shufflevector(p0, p1, 0,1,2,3,4,5,6,7,8,9,10,11,12,13,14,15);
        }
        {
            v8h m10 = *(const v8h*)(gm[1] + k0 + ka);
            v8h m11 = *(const v8h*)(gm[1] + k0 + ka + 16);
            v8h j10 = *(const v8h*)(gj[1] + k0 + ka);
            v8h j11 = *(const v8h*)(gj[1] + k0 + ka + 16);
            v8h p0 = m10 * j10;
            v8h p1 = m11 * j11;
            a1 = __builtin_shufflevector(p0, p1, 0,1,2,3,4,5,6,7,8,9,10,11,12,13,14,15);
        }
        v16h b0 = load_fragB(Wr0, k0, kb);
        v16h b1 = load_fragB(Wr1, k0, kb);
        acc[0][0] = __builtin_amdgcn_wmma_f32_16x16x32_f16(false, a0, false, b0, (short)0, acc[0][0], false, false);
        acc[0][1] = __builtin_amdgcn_wmma_f32_16x16x32_f16(false, a0, false, b1, (short)0, acc[0][1], false, false);
        acc[1][0] = __builtin_amdgcn_wmma_f32_16x16x32_f16(false, a1, false, b0, (short)0, acc[1][0], false, false);
        acc[1][1] = __builtin_amdgcn_wmma_f32_16x16x32_f16(false, a1, false, b1, (short)0, acc[1][1], false, false);
    }

#pragma unroll
    for (int rf = 0; rf < 2; ++rf) {
        const int mbase = m0 + rf * 16 + ((lane >> 4) << 3);
#pragma unroll
        for (int cf = 0; cf < 2; ++cf) {
            const int col = n0 + cf * 16 + (lane & 15);
            const bool colOK = col < HS;
            const float bv = colOK ? (pb1[col] + genC[col]) : 0.0f;
#pragma unroll
            for (int v = 0; v < 8; ++v) {
                int mko = mbase + v;
                int m2 = mko / KK, k2 = mko - m2 * KK;
                int j2 = m2 - KK + k2; if (j2 < 0) j2 = 0;
                _Float16 outv = (_Float16)0.0f;
                if (colOK) {
                    int dbin = binize(enk[m2] - stk[j2]);
                    int si = spkk[m2], sj = spkk[j2];
                    int lab = (si == 0 || sj == 0) ? 0 : (si == sj ? 1 : 2);
                    float r = acc[rf][cf][v]
                            + t_i[(size_t)m2 * HS + col] + t_j[(size_t)j2 * HS + col]
                            + distC[dbin * HS + col] + spkC[lab * HS + col] + bv;
                    outv = (_Float16)fmaxf(r, 0.0f);
                }
                h1P[(size_t)mko * KH + col] = outv;
            }
        }
    }
}

// Final: sp = h2@w3 + b3 ; sij = sk[m]+sk[j]+sp (masked) ; zero last column.
__global__ void pair_out_kernel(
    const float* __restrict__ h2, const float* __restrict__ w3,
    const float* __restrict__ b3, const float* __restrict__ skv,
    float* __restrict__ out)
{
    int idx = blockIdx.x * blockDim.x + threadIdx.x;
    if (idx >= MK) return;
    int m = idx / KK, k = idx - m * KK;
    int jraw = m - KK + k;
    int j = jraw < 0 ? 0 : jraw;
    float sp = b3[0];
    const float* hrow = h2 + (size_t)idx * HS;
#pragma unroll 5
    for (int d = 0; d < HS; ++d) sp += hrow[d] * w3[d];
    float v = (jraw >= 0) ? (skv[m] + skv[j] + sp) : -1.0e9f;
    out[(size_t)m * (KK + 1) + k] = v;
    if (k == 0) out[(size_t)m * (KK + 1) + KK] = 0.0f;
}

static inline void pack(const float* src, int rows, int cols, int ld,
                        _Float16* dst, int Rp, int Cp, hipStream_t s) {
    int n = Rp * Cp;
    pack_f16_kernel<<<(n + 255) / 256, 256, 0, s>>>(src, rows, cols, ld, dst, Rp, Cp);
}

extern "C" void kernel_launch(void* const* d_in, const int* in_sizes, int n_in,
                              void* d_out, int out_size, void* d_ws, size_t ws_size,
                              hipStream_t stream) {
    const float* x        = (const float*)d_in[0];
    const float* w_ih     = (const float*)d_in[1];
    const float* w_hh     = (const float*)d_in[2];
    const float* lb       = (const float*)d_in[3];
    const float* attn_w1  = (const float*)d_in[4];
    const float* attn_b1  = (const float*)d_in[5];
    const float* attn_w2  = (const float*)d_in[6];
    const float* attn_b2  = (const float*)d_in[7];
    const float* attn_w3  = (const float*)d_in[8];
    const float* attn_b3  = (const float*)d_in[9];
    const float* ment_w1  = (const float*)d_in[10];
    const float* ment_b1  = (const float*)d_in[11];
    const float* ment_w2  = (const float*)d_in[12];
    const float* ment_b2  = (const float*)d_in[13];
    const float* ment_w3  = (const float*)d_in[14];
    const float* ment_b3  = (const float*)d_in[15];
    const float* pair_w1  = (const float*)d_in[16];
    const float* pair_b1  = (const float*)d_in[17];
    const float* pair_w2  = (const float*)d_in[18];
    const float* pair_b2  = (const float*)d_in[19];
    const float* pair_w3  = (const float*)d_in[20];
    const float* pair_b3  = (const float*)d_in[21];
    const float* width_tab   = (const float*)d_in[22];
    const float* dist_tab    = (const float*)d_in[23];
    const float* genre_tab   = (const float*)d_in[24];
    const float* speaker_tab = (const float*)d_in[25];
    const int*   starts      = (const int*)d_in[26];
    const int*   ends        = (const int*)d_in[27];
    const int*   speaker_ids = (const int*)d_in[28];
    const int*   genre_p     = (const int*)d_in[29];

    size_t off = 0;
    auto alignup = [](size_t v, size_t a) { return (v + a - 1) / a * a; };
    auto allocB = [&](size_t bytes) -> void* {
        off = alignup(off, 64);
        void* p = (char*)d_ws + off;
        off += bytes;
        return p;
    };
    auto allocF = [&](size_t n) { return (float*)allocB(n * sizeof(float)); };
    auto allocH = [&](size_t n) { return (_Float16*)allocB(n * sizeof(_Float16)); };
    auto allocI = [&](size_t n) { return (int*)allocB(n * sizeof(int)); };

    float* pre0    = allocF((size_t)TT * G4H);
    float* pre1    = allocF((size_t)TT * G4H);
    float* statesA = allocF((size_t)TT * 2 * HH);
    float* statesB = allocF((size_t)TT * 2 * HH);
    float* attn_s  = allocF(TT);
    float* gbuf    = allocF((size_t)SS * DD);     // re-used for ph1P/ph2 later
    float* smv     = allocF(SS);
    float* skv     = allocF(MM);
    float* t_i     = allocF((size_t)MM2 * HS);
    float* t_j     = allocF((size_t)MM2 * HS);
    float* distC   = allocF(9 * HS);
    float* genC    = allocF(HS);
    float* spkC    = allocF(3 * HS);

    _Float16* xP    = allocH((size_t)TT * KIN);
    _Float16* wihP0 = allocH((size_t)G4H * KIN);
    _Float16* wihP1 = allocH((size_t)G4H * KIN);
    _Float16* stAP  = allocH((size_t)TT * KIN);
    _Float16* stBP  = allocH((size_t)TT * KIN);
    _Float16* aw1P  = allocH((size_t)KH * KIN);
    _Float16* aw2P  = allocH((size_t)KH * KH);
    _Float16* aw3P  = allocH((size_t)32 * KH);
    _Float16* mw1P  = allocH((size_t)KH * KD);
    _Float16* mw2P  = allocH((size_t)KH * KH);
    _Float16* mw3P  = allocH((size_t)32 * KH);
    _Float16* w1iP  = allocH((size_t)KH * KD);
    _Float16* w1jP  = allocH((size_t)KH * KD);
    _Float16* w1pP  = allocH((size_t)KH * KD);
    _Float16* pw2P  = allocH((size_t)KH * KH);
    _Float16* ah1P  = allocH((size_t)TT * KH);
    _Float16* ah2P  = allocH((size_t)TT * KH);
    _Float16* gP    = allocH((size_t)SS * KD);
    _Float16* mh1P  = allocH((size_t)SS * KH);
    _Float16* mh2P  = allocH((size_t)SS * KH);
    _Float16* gkP   = allocH((size_t)MM2 * KD);

    int* flag = allocI(SS);
    int* keep = allocI(MM);
    int* stk  = allocI(MM2);
    int* enk  = allocI(MM2);
    int* spkk = allocI(MM2);

    // pair activations alias gbuf (dead after gather): fp32 ph2 then f16 ph1P
    float*    ph2  = gbuf;                                  // MK*HS fp32
    _Float16* ph1P = (_Float16*)(gbuf + (size_t)MK * HS);   // MKP*KH f16

    float* out = (float*)d_out;

    // --- packs (weights + x) ---
    pack(x, TT, EE, EE, xP, TT, KIN, stream);
    pack(attn_w1, HS, 2 * HH, 2 * HH, aw1P, KH, KIN, stream);
    pack(attn_w2, HS, HS, HS, aw2P, KH, KH, stream);
    pack(attn_w3, 1, HS, HS, aw3P, 32, KH, stream);
    pack(ment_w1, HS, DD, DD, mw1P, KH, KD, stream);
    pack(ment_w2, HS, HS, HS, mw2P, KH, KH, stream);
    pack(ment_w3, 1, HS, HS, mw3P, 32, KH, stream);
    pack(pair_w1,          HS, DD, PP, w1iP, KH, KD, stream);
    pack(pair_w1 + DD,     HS, DD, PP, w1jP, KH, KD, stream);
    pack(pair_w1 + 2 * DD, HS, DD, PP, w1pP, KH, KD, stream);
    pack(pair_w2, HS, HS, HS, pw2P, KH, KH, stream);
    meta_tab_kernel<<<1, 256, 0, stream>>>(pair_w1, dist_tab, genre_tab, genre_p,
                                           speaker_tab, distC, genC, spkC);

    // --- 2-layer biLSTM ---
    const _Float16* inpP = xP;
    float* lay_out[2] = { statesA, statesB };
    _Float16* lay_pk[2] = { stAP, stBP };
    for (int l = 0; l < 2; ++l) {
        const float* wih0 = w_ih + (size_t)(l * 2 + 0) * G4H * EE;
        const float* wih1 = w_ih + (size_t)(l * 2 + 1) * G4H * EE;
        const float* whh0 = w_hh + (size_t)(l * 2 + 0) * G4H * HH;
        const float* whh1 = w_hh + (size_t)(l * 2 + 1) * G4H * HH;
        const float* b0   = lb + (size_t)(l * 2 + 0) * G4H;
        const float* b1v  = lb + (size_t)(l * 2 + 1) * G4H;
        pack(wih0, G4H, EE, EE, wihP0, G4H, KIN, stream);
        pack(wih1, G4H, EE, EE, wihP1, G4H, KIN, stream);
        gemm_pk_kernel<<<dim3(TT / 32, G4H / 32), 32, 0, stream>>>(
            inpP, KIN, wihP0, KIN, b0, KIN, TT, G4H, pre0, G4H, nullptr, 0, 0);
        gemm_pk_kernel<<<dim3(TT / 32, G4H / 32), 32, 0, stream>>>(
            inpP, KIN, wihP1, KIN, b1v, KIN, TT, G4H, pre1, G4H, nullptr, 0, 0);
        lstm_scan_kernel<<<2, G4H, 0, stream>>>(pre0, pre1, whh0, whh1, lay_out[l]);
        pack(lay_out[l], TT, 2 * HH, 2 * HH, lay_pk[l], TT, KIN, stream);
        inpP = lay_pk[l];
    }

    // --- attention MLP -> attn scores ---
    gemm_pk_kernel<<<dim3(TT / 32, KH / 32), 32, 0, stream>>>(
        stBP, KIN, aw1P, KIN, attn_b1, KIN, TT, HS, nullptr, 0, ah1P, KH, 1);
    gemm_pk_kernel<<<dim3(TT / 32, KH / 32), 32, 0, stream>>>(
        ah1P, KH, aw2P, KH, attn_b2, KH, TT, HS, nullptr, 0, ah2P, KH, 1);
    gemm_pk_kernel<<<dim3(TT / 32, 1), 32, 0, stream>>>(
        ah2P, KH, aw3P, KH, attn_b3, KH, TT, 1, attn_s, 1, nullptr, 0, 0);

    // --- span features + pack ---
    span_kernel<<<SS, 128, 0, stream>>>(x, statesB, attn_s, starts, ends,
                                        width_tab, gbuf);
    pack(gbuf, SS, DD, DD, gP, SS, KD, stream);

    // --- mention MLP -> sm ---
    gemm_pk_kernel<<<dim3(SS / 32, KH / 32), 32, 0, stream>>>(
        gP, KD, mw1P, KD, ment_b1, KD, SS, HS, nullptr, 0, mh1P, KH, 1);
    gemm_pk_kernel<<<dim3(SS / 32, KH / 32), 32, 0, stream>>>(
        mh1P, KH, mw2P, KH, ment_b2, KH, SS, HS, nullptr, 0, mh2P, KH, 1);
    gemm_pk_kernel<<<dim3(SS / 32, 1), 32, 0, stream>>>(
        mh2P, KH, mw3P, KH, ment_b3, KH, SS, 1, smv, 1, nullptr, 0, 0);

    // --- top-300, compact, gather ---
    rank_kernel<<<(SS + 255) / 256, 256, 0, stream>>>(smv, flag);
    compact_kernel<<<(SS + 255) / 256, 256, 0, stream>>>(flag, keep);
    gather_kernel<<<MM2, 128, 0, stream>>>(gbuf, smv, starts, ends, speaker_ids,
                                           keep, gkP, skv, stk, enk, spkk);

    // --- pair head ---
    gemm_pk_kernel<<<dim3(MM2 / 32, KH / 32), 32, 0, stream>>>(
        gkP, KD, w1iP, KD, nullptr, KD, MM2, HS, t_i, HS, nullptr, 0, 0);
    gemm_pk_kernel<<<dim3(MM2 / 32, KH / 32), 32, 0, stream>>>(
        gkP, KD, w1jP, KD, nullptr, KD, MM2, HS, t_j, HS, nullptr, 0, 0);

    pair_h1_kernel<<<dim3(MKP / 32, KH / 32), 32, 0, stream>>>(
        gkP, w1pP, pair_b1, t_i, t_j, stk, enk, spkk, distC, genC, spkC, ph1P);

    gemm_pk_kernel<<<dim3(MKP / 32, KH / 32), 32, 0, stream>>>(
        ph1P, KH, pw2P, KH, pair_b2, KH, MK, HS, ph2, HS, nullptr, 0, 1);

    pair_out_kernel<<<(MK + 255) / 256, 256, 0, stream>>>(
        ph2, pair_w3, pair_b3, skv, out);
}
